// HybridGNNModel_78649441124916
// MI455X (gfx1250) — matrix operand order repaired and checked by reading, not compiled, not run on hardware
//
#include <hip/hip_runtime.h>
#include <hip/hip_bf16.h>
#include <math.h>

#define NNODES 50000
#define NEDGES 600000
#define INDIM  256
#define H1DIM  128
#define H2DIM  64
#define H3DIM  64
#define NHEADS 8
#define ETOT   (NEDGES + NNODES)
#define GATF   (NHEADS * H2DIM)   // 512
#define KMAX   256                // largest GEMM K (fits LDS panel: 256*64*2B = 32KB)

typedef __attribute__((ext_vector_type(16))) __bf16 v16bf;
typedef __attribute__((ext_vector_type(8)))  float  v8f;

// ---------- order-preserving float<->uint mapping for atomicMax on floats ----------
__device__ __forceinline__ unsigned f2ord(float f) {
  unsigned u = __float_as_uint(f);
  return (u & 0x80000000u) ? ~u : (u | 0x80000000u);
}
__device__ __forceinline__ float ord2f(unsigned u) {
  return __uint_as_float((u & 0x80000000u) ? (u ^ 0x80000000u) : ~u);
}
#define ORD_NEG_INF 0x007FFFFFu  // f2ord(-inf)

__device__ __forceinline__ unsigned short f2bf(float f) {
  __bf16 h = (__bf16)f;
  return __builtin_bit_cast(unsigned short, h);
}

// ---------------------------- utility fills ----------------------------
__global__ void fill_f32(float* __restrict__ p, float v, int n) {
  int i = blockIdx.x * blockDim.x + threadIdx.x;
  if (i < n) p[i] = v;
}
__global__ void fill_u32(unsigned* __restrict__ p, unsigned v, int n) {
  int i = blockIdx.x * blockDim.x + threadIdx.x;
  if (i < n) p[i] = v;
}

// ---------------------------- degree / dinv ----------------------------
__global__ void deg_scatter(float* __restrict__ deg, const int* __restrict__ dst,
                            const float* __restrict__ ew, int e) {
  int i = blockIdx.x * blockDim.x + threadIdx.x;
  if (i < e) atomicAdd(&deg[dst[i]], ew[i]);
}
__global__ void deg_to_dinv(float* __restrict__ d, int n) {
  int i = blockIdx.x * blockDim.x + threadIdx.x;
  if (i < n) { float v = d[i]; d[i] = (v > 0.f) ? rsqrtf(v) : 0.f; }
}

// ------------------- bf16 WMMA GEMM: C[M,Nc] = A[M,K] @ B[K,Nc] -------------------
// Workgroup macro-tile 128x64: 8 waves, each wave owns 16 rows x 64 cols
// (4 accumulators -> 4 WMMAs per K-step, A fragment reused across them).
// B panel (K x 64) is converted f32->bf16 once into LDS, column-major, so each
// B fragment is one contiguous 32B ds_load per lane.
__global__ __launch_bounds__(256) void gemm_wmma_bf16(
    const float* __restrict__ A, const float* __restrict__ B, float* __restrict__ C,
    int M, int K, int Nc) {
  __shared__ unsigned short Bl[KMAX * 64];   // [col 0..63][k 0..K-1], bf16

  const int tid  = threadIdx.x;
  const int lane = tid & 31;
  const int wave = tid >> 5;
  const int tiles_m  = M >> 4;
  const int blocks_n = Nc >> 6;              // 64-col panels
  const int bm = blockIdx.x / blocks_n;
  const int bn = blockIdx.x % blocks_n;

  // ---- cooperative LDS fill: B[:, bn*64 .. bn*64+63] -> bf16, column-major ----
  {
    const int khalfK = K >> 1;               // pairs along k
    unsigned* Bu = (unsigned*)Bl;
    for (int i = tid; i < 32 * K; i += 256) {   // 64*K/2 pairs
      const int c  = i / khalfK;
      const int k2 = i % khalfK;
      const int gcol = bn * 64 + c;
      const unsigned lo = f2bf(B[(size_t)(2 * k2)     * Nc + gcol]);
      const unsigned hi = f2bf(B[(size_t)(2 * k2 + 1) * Nc + gcol]);
      Bu[c * khalfK + k2] = lo | (hi << 16);
    }
  }
  __syncthreads();

  const int tm = bm * 8 + wave;              // this wave's 16-row tile
  if (tm >= tiles_m) return;                 // uniform per wave; after barrier

  const int r  = lane & 15;
  const int kh = lane >> 4;
  const int arow = tm * 16 + r;

  v8f acc0 = {}, acc1 = {}, acc2 = {}, acc3 = {};
  for (int k0 = 0; k0 < K; k0 += 32) {
    // A fragment (16x32 bf16): lane half kh holds K [16kh..16kh+15] contiguous
    v16bf af;
    const float* ap = A + (size_t)arow * K + k0 + kh * 16;
    if (k0 + 32 < K) __builtin_prefetch(ap + 32, 0, 0);   // global_prefetch_b8
#pragma unroll
    for (int e = 0; e < 8; ++e) {
      af[e]     = (__bf16)ap[e];
      af[8 + e] = (__bf16)ap[8 + e];
    }
    // 4 N-tiles: B fragment = contiguous 16 bf16 from LDS (col-major panel)
#pragma unroll
    for (int j = 0; j < 4; ++j) {
      const int c = j * 16 + r;              // local panel column
      const v16bf bf = *reinterpret_cast<const v16bf*>(&Bl[c * K + k0 + 16 * kh]);
      v8f* accp = (j == 0) ? &acc0 : (j == 1) ? &acc1 : (j == 2) ? &acc2 : &acc3;
      *accp = __builtin_amdgcn_wmma_f32_16x16x32_bf16(
          false, af, false, bf, (short)0, *accp, false, false);
    }
  }

  // store: C/D layout -> row rr + 8*kh, col = lane&15 within each 16x16 tile
  const size_t crow0 = (size_t)(tm * 16 + 8 * kh) * Nc + bn * 64 + r;
#pragma unroll
  for (int rr = 0; rr < 8; ++rr) {
    C[crow0 + (size_t)rr * Nc +  0] = acc0[rr];
    C[crow0 + (size_t)rr * Nc + 16] = acc1[rr];
    C[crow0 + (size_t)rr * Nc + 32] = acc2[rr];
    C[crow0 + (size_t)rr * Nc + 48] = acc3[rr];
  }
}

// ---------------------------- GCN scatter-add ----------------------------
__global__ void gcn_scatter(const float* __restrict__ h, const float* __restrict__ dinv,
                            const int* __restrict__ src, const int* __restrict__ dst,
                            const float* __restrict__ ew, float* __restrict__ acc, int F) {
  int t = blockIdx.x * blockDim.x + threadIdx.x;
  if (t >= ETOT) return;
  int s_, d_; float w;
  if (t < NEDGES) { s_ = src[t]; d_ = dst[t]; w = ew[t]; }
  else            { s_ = d_ = t - NEDGES;     w = 1.0f; }
  const float coeff = dinv[s_] * w * dinv[d_];
  const float* hs = h + (size_t)s_ * F;
  float* ad = acc + (size_t)d_ * F;
  for (int c = 0; c < F; c += 4) {
    float4 v = *(const float4*)(hs + c);
    atomicAdd(ad + c + 0, coeff * v.x);
    atomicAdd(ad + c + 1, coeff * v.y);
    atomicAdd(ad + c + 2, coeff * v.z);
    atomicAdd(ad + c + 3, coeff * v.w);
  }
}

__global__ void bias_relu(const float* __restrict__ acc, const float* __restrict__ b,
                          float* __restrict__ out, int total, int F) {
  int i = blockIdx.x * blockDim.x + threadIdx.x;
  if (i < total) { float v = acc[i] + b[i % F]; out[i] = v > 0.f ? v : 0.f; }
}

// ---------------------------- GAT pieces ----------------------------
__global__ void att_dots(const float* __restrict__ h2, const float* __restrict__ att,
                         float* __restrict__ out) {
  int i = blockIdx.x * blockDim.x + threadIdx.x;
  if (i >= NNODES * NHEADS) return;
  const int n = i >> 3, hd = i & 7;
  const float* hp = h2 + (size_t)n * GATF + hd * H2DIM;
  const float* ap = att + hd * H2DIM;
  float s = 0.f;
#pragma unroll 8
  for (int c = 0; c < H2DIM; ++c) s += hp[c] * ap[c];
  out[i] = s;
}

__global__ void gat_edge_logits(const float* __restrict__ as, const float* __restrict__ ad,
                                const int* __restrict__ src, const int* __restrict__ dst,
                                float* __restrict__ p, unsigned* __restrict__ mx) {
  int i = blockIdx.x * blockDim.x + threadIdx.x;
  if (i >= ETOT * NHEADS) return;
  const int t = i >> 3, hd = i & 7;
  int s_, d_;
  if (t < NEDGES) { s_ = src[t]; d_ = dst[t]; } else { s_ = d_ = t - NEDGES; }
  float e = as[s_ * NHEADS + hd] + ad[d_ * NHEADS + hd];
  e = e > 0.f ? e : 0.2f * e;                       // leaky_relu(0.2)
  p[i] = e;
  atomicMax(&mx[d_ * NHEADS + hd], f2ord(e));
}

__global__ void gat_edge_exp(const int* __restrict__ dst, float* __restrict__ p,
                             const unsigned* __restrict__ mx, float* __restrict__ s) {
  int i = blockIdx.x * blockDim.x + threadIdx.x;
  if (i >= ETOT * NHEADS) return;
  const int t = i >> 3, hd = i & 7;
  const int d_ = (t < NEDGES) ? dst[t] : t - NEDGES;
  float pv = __expf(p[i] - ord2f(mx[d_ * NHEADS + hd]));
  p[i] = pv;
  atomicAdd(&s[d_ * NHEADS + hd], pv);
}

__global__ void gat_scatter(const float* __restrict__ h2, const int* __restrict__ src,
                            const int* __restrict__ dst, const float* __restrict__ p,
                            const float* __restrict__ s, float* __restrict__ acc) {
  int i = blockIdx.x * blockDim.x + threadIdx.x;
  if (i >= ETOT * NHEADS) return;
  const int t = i >> 3, hd = i & 7;
  int s_, d_;
  if (t < NEDGES) { s_ = src[t]; d_ = dst[t]; } else { s_ = d_ = t - NEDGES; }
  const float alpha = p[i] / (s[d_ * NHEADS + hd] + 1e-16f);
  const float* hp = h2 + (size_t)s_ * GATF + hd * H2DIM;
  float* ap = acc + (size_t)d_ * GATF + hd * H2DIM;
  for (int c = 0; c < H2DIM; c += 4) {
    float4 v = *(const float4*)(hp + c);
    atomicAdd(ap + c + 0, alpha * v.x);
    atomicAdd(ap + c + 1, alpha * v.y);
    atomicAdd(ap + c + 2, alpha * v.z);
    atomicAdd(ap + c + 3, alpha * v.w);
  }
}

__global__ void gat_finalize(const float* __restrict__ acc, const float* __restrict__ b,
                             float* __restrict__ out) {
  int i = blockIdx.x * blockDim.x + threadIdx.x;
  if (i >= NNODES * H2DIM) return;
  const int n = i / H2DIM, c = i % H2DIM;
  float s = 0.f;
#pragma unroll
  for (int hd = 0; hd < NHEADS; ++hd) s += acc[(size_t)n * GATF + hd * H2DIM + c];
  float v = s * (1.0f / NHEADS) + b[c];
  out[i] = v > 0.f ? v : 0.f;
}

// ---------------------------- final FC ----------------------------
__global__ void fc_kernel(const float* __restrict__ h, const float* __restrict__ w,
                          const float* __restrict__ b, float* __restrict__ out) {
  int n = blockIdx.x * blockDim.x + threadIdx.x;
  if (n >= NNODES) return;
  float s = 0.f;
#pragma unroll 8
  for (int c = 0; c < H3DIM; ++c) s += h[(size_t)n * H3DIM + c] * w[c];
  out[n] = s + b[0];
}

// ---------------------------- launcher ----------------------------
extern "C" void kernel_launch(void* const* d_in, const int* in_sizes, int n_in,
                              void* d_out, int out_size, void* d_ws, size_t ws_size,
                              hipStream_t stream) {
  (void)in_sizes; (void)n_in; (void)out_size; (void)ws_size;
  const float* x       = (const float*)d_in[0];
  const int*   ei      = (const int*)  d_in[1];
  const float* ew      = (const float*)d_in[2];
  const float* gc1_w   = (const float*)d_in[3];
  const float* gc1_b   = (const float*)d_in[4];
  const float* gat_w   = (const float*)d_in[5];
  const float* att_src = (const float*)d_in[6];
  const float* att_dst = (const float*)d_in[7];
  const float* gat_b   = (const float*)d_in[8];
  const float* gc2_w   = (const float*)d_in[9];
  const float* gc2_b   = (const float*)d_in[10];
  const float* fc_w    = (const float*)d_in[11];
  const float* fc_b    = (const float*)d_in[12];
  const int* esrc = ei;
  const int* edst = ei + NEDGES;
  float* out = (float*)d_out;

  // workspace layout (floats)
  float* W = (float*)d_ws;
  size_t o = 0;
  float*    dinv = W + o; o += NNODES;
  float*    Hb   = W + o; o += (size_t)NNODES * GATF;   // GEMM outputs (max N x 512)
  float*    Acc  = W + o; o += (size_t)NNODES * GATF;   // scatter accumulators
  float*    Lout = W + o; o += (size_t)NNODES * H1DIM;  // layer outputs (max N x 128)
  float*    As   = W + o; o += (size_t)NNODES * NHEADS;
  float*    Ad   = W + o; o += (size_t)NNODES * NHEADS;
  unsigned* Mx   = (unsigned*)(W + o); o += (size_t)NNODES * NHEADS;
  float*    S    = W + o; o += (size_t)NNODES * NHEADS;
  float*    P    = W + o; o += (size_t)ETOT * NHEADS;

  const int T = 256;
  auto nb = [](long n) { return (int)((n + 255) / 256); };
  const int tiles_m  = NNODES / 16;                 // 3125
  const int blocks_m = (tiles_m + 7) / 8;           // 391

  // ---- shared symmetric-norm coefficients (self-loop weight 1 -> init deg=1) ----
  fill_f32<<<nb(NNODES), T, 0, stream>>>(dinv, 1.0f, NNODES);
  deg_scatter<<<nb(NEDGES), T, 0, stream>>>(dinv, edst, ew, NEDGES);
  deg_to_dinv<<<nb(NNODES), T, 0, stream>>>(dinv, NNODES);

  // ---- GCN layer 1: 256 -> 128 ----
  {
    gemm_wmma_bf16<<<blocks_m * (H1DIM / 64), 256, 0, stream>>>(x, gc1_w, Hb, NNODES, INDIM, H1DIM);
    fill_f32<<<nb((long)NNODES * H1DIM), T, 0, stream>>>(Acc, 0.f, NNODES * H1DIM);
    gcn_scatter<<<nb(ETOT), T, 0, stream>>>(Hb, dinv, esrc, edst, ew, Acc, H1DIM);
    bias_relu<<<nb((long)NNODES * H1DIM), T, 0, stream>>>(Acc, gc1_b, Lout, NNODES * H1DIM, H1DIM);
  }

  // ---- GAT layer: 128 -> 8 heads x 64, mean ----
  {
    gemm_wmma_bf16<<<blocks_m * (GATF / 64), 256, 0, stream>>>(Lout, gat_w, Hb, NNODES, H1DIM, GATF);
    att_dots<<<nb((long)NNODES * NHEADS), T, 0, stream>>>(Hb, att_src, As);
    att_dots<<<nb((long)NNODES * NHEADS), T, 0, stream>>>(Hb, att_dst, Ad);
    fill_u32<<<nb((long)NNODES * NHEADS), T, 0, stream>>>(Mx, ORD_NEG_INF, NNODES * NHEADS);
    gat_edge_logits<<<nb((long)ETOT * NHEADS), T, 0, stream>>>(As, Ad, esrc, edst, P, Mx);
    fill_f32<<<nb((long)NNODES * NHEADS), T, 0, stream>>>(S, 0.f, NNODES * NHEADS);
    gat_edge_exp<<<nb((long)ETOT * NHEADS), T, 0, stream>>>(edst, P, Mx, S);
    fill_f32<<<nb((long)NNODES * GATF), T, 0, stream>>>(Acc, 0.f, NNODES * GATF);
    gat_scatter<<<nb((long)ETOT * NHEADS), T, 0, stream>>>(Hb, esrc, edst, P, S, Acc);
    gat_finalize<<<nb((long)NNODES * H2DIM), T, 0, stream>>>(Acc, gat_b, Lout);
  }

  // ---- GCN layer 2: 64 -> 64 ----
  {
    gemm_wmma_bf16<<<blocks_m * (H3DIM / 64), 256, 0, stream>>>(Lout, gc2_w, Hb, NNODES, H2DIM, H3DIM);
    fill_f32<<<nb((long)NNODES * H3DIM), T, 0, stream>>>(Acc, 0.f, NNODES * H3DIM);
    gcn_scatter<<<nb(ETOT), T, 0, stream>>>(Hb, dinv, esrc, edst, ew, Acc, H3DIM);
    bias_relu<<<nb((long)NNODES * H3DIM), T, 0, stream>>>(Acc, gc2_b, Lout, NNODES * H3DIM, H3DIM);
  }

  // ---- final FC: 64 -> 1 ----
  fc_kernel<<<nb(NNODES), T, 0, stream>>>(Lout, fc_w, fc_b, out);
}
